// Vector_Attention_86835648791014
// MI455X (gfx1250) — matrix operand order repaired and checked by reading, not compiled, not run on hardware
//
#include <hip/hip_runtime.h>
#include <math.h>

static constexpr int kN   = 512;
static constexpr int kD   = 64;
static constexpr int kLXY = 24;
static constexpr int kLZ  = 16;

typedef __attribute__((ext_vector_type(16))) __bf16       bf16x16;
typedef __attribute__((ext_vector_type(8)))  float        f32x8;
typedef __attribute__((ext_vector_type(4)))  float        f32x4;
typedef __attribute__((ext_vector_type(4)))  unsigned int u32x4;

#if defined(__has_builtin)
#  if __has_builtin(__builtin_amdgcn_tensor_load_to_lds) && \
      __has_builtin(__builtin_amdgcn_s_wait_tensorcnt)
#    define USE_TDM 1
#  endif
#endif
#ifndef USE_TDM
#  define USE_TDM 0
#endif

union Frag {
  u32x4          q4[2];
  unsigned short us[16];
  bf16x16        v;
};

__device__ __forceinline__ unsigned short f2bf(float f) {
  union { __bf16 b; unsigned short u; } x;
  x.b = (__bf16)f;                                   // native v_cvt on gfx1250
  return x.u;
}
__device__ __forceinline__ float bf2f(unsigned short h) {
  union { unsigned u; float f; } x; x.u = ((unsigned)h) << 16;
  return x.f;
}

#if USE_TDM
typedef __attribute__((ext_vector_type(8))) int i32x8;
typedef __attribute__((ext_vector_type(4))) int i32x4;

// DMA a 2-D f32 tile (tile_rows x 64, row stride 64) from global into LDS.
__device__ __forceinline__ void tdm_load_tile_f32(unsigned lds_addr,
                                                  const float* gptr,
                                                  unsigned tile_rows) {
  unsigned long long ga = (unsigned long long)gptr;
  u32x4 g0;
  g0[0] = 1u;                                        // count=1, user descriptor
  g0[1] = lds_addr;                                  // LDS byte address
  g0[2] = (unsigned)(ga & 0xFFFFFFFFu);              // global_addr[31:0]
  g0[3] = (unsigned)((ga >> 32) & 0x1FFFFFFu) | (2u << 30);  // addr[56:32], type=2
  i32x8 g1;
  g1[0] = (int)(2u << 16);                           // data_size=4B, no multicast
  g1[1] = (int)((kD & 0xFFFFu) << 16);               // tensor_dim0 = 64 (lo16)
  g1[2] = (int)((kN & 0xFFFFu) << 16);               // dim0 hi=0 | tensor_dim1=512
  g1[3] = (int)((kD & 0xFFFFu) << 16);               // dim1 hi=0 | tile_dim0=64
  g1[4] = (int)(tile_rows & 0xFFFFu);                // tile_dim1 | tile_dim2=0
  g1[5] = (int)kD;                                   // tensor_dim0_stride = 64
  g1[6] = 0;
  g1[7] = 0;
  i32x4 z4 = {0, 0, 0, 0};
#if defined(__clang_major__) && (__clang_major__ >= 23)
  i32x8 z8 = {0, 0, 0, 0, 0, 0, 0, 0};
  __builtin_amdgcn_tensor_load_to_lds(g0, g1, z4, z4, z8, 0);
#else
  __builtin_amdgcn_tensor_load_to_lds(g0, g1, z4, z4, 0);
#endif
}
__device__ __forceinline__ unsigned lds_off(const void* p) {
  return (unsigned)(unsigned long long)p;            // low 32 bits = LDS offset
}
#endif

// ---------------- K1a: qkv = feat @ W_qkv + b ----------------
__global__ __launch_bounds__(256) void qkv_kernel(
    const float* __restrict__ feat, const float* __restrict__ Wqkv,
    const float* __restrict__ bqkv, float* __restrict__ q,
    float* __restrict__ k, float* __restrict__ v) {
  int gid = blockIdx.x * 256 + threadIdx.x;          // 512*192 threads
  int i = gid / 192, oc = gid % 192;
  float acc = bqkv[oc];
  const float* fr = feat + i * kD;
#pragma unroll 8
  for (int kk = 0; kk < kD; ++kk) acc = fmaf(fr[kk], Wqkv[kk * 192 + oc], acc);
  float* dst = (oc < 64) ? q : (oc < 128) ? k : v;
  dst[i * kD + (oc & 63)] = acc;
}

// ---------------- K1b: bf16 transposed MLP weights ----------------
__global__ __launch_bounds__(256) void prep_weights(
    const float* __restrict__ Wg1, const float* __restrict__ Wg2,
    unsigned short* __restrict__ WT1, unsigned short* __restrict__ WT2) {
  for (int e = threadIdx.x; e < kD * kD; e += 256) {
    int n = e >> 6, kk = e & 63;                     // WT[n][k] = W[k][n]
    WT1[e] = f2bf(Wg1[kk * kD + n]);
    WT2[e] = f2bf(Wg2[kk * kD + n]);
  }
}

// ---------------- K2: pairwise MLP + encodings + per-channel softmax ----------------
__global__ __launch_bounds__(128) void vec_att_kernel(
    const float* __restrict__ q, const float* __restrict__ k,
    const float* __restrict__ v, const float* __restrict__ dist,
    const float* __restrict__ bg1, const float* __restrict__ bg2,
    const unsigned short* __restrict__ WT1, const unsigned short* __restrict__ WT2,
    const float* __restrict__ tx, const float* __restrict__ ty,
    const float* __restrict__ tz, float* __restrict__ att) {
  __shared__ unsigned short sTX[3 * kLXY * kD];      //  9216 B
  __shared__ unsigned short sTY[3 * kLXY * kD];      //  9216 B
  __shared__ unsigned short sTZ[3 * kLZ * kD];       //  6144 B
  __shared__ float          sQf[16 * kD];            //  4096 B
  __shared__ float          sKf[2][16 * kD];         //  8192 B (double buffered)
  __shared__ float          sVf[2][16 * kD];         //  8192 B
  __shared__ unsigned short sH[128 * kD];            // 16384 B (half of the pair tile)
  __shared__ unsigned       sIDX[256];               //  1024 B  (total 61.0 KB)

  const int tid  = threadIdx.x;
  const int l    = tid & 31;
  const int w    = tid >> 5;
  const int ln   = l & 15;
  const int half = l >> 4;
  const int ti   = blockIdx.x;

  // Stage tables (bf16).
  for (int e = tid; e < 3 * kLXY * kD; e += 128) sTX[e] = f2bf(tx[e]);
  for (int e = tid; e < 3 * kLXY * kD; e += 128) sTY[e] = f2bf(ty[e]);
  for (int e = tid; e < 3 * kLZ * kD; e += 128)  sTZ[e] = f2bf(tz[e]);

#if USE_TDM
  if (w == 0) {                                      // TDM prologue: q-tile + j-tile 0
    tdm_load_tile_f32(lds_off(&sQf[0]),    q + ti * 16 * kD, 16);
    tdm_load_tile_f32(lds_off(&sKf[0][0]), k, 16);
    tdm_load_tile_f32(lds_off(&sVf[0][0]), v, 16);
  }
#else
  for (int e = tid; e < 16 * kD; e += 128) {
    int r = e >> 6, c = e & 63;
    sQf[e] = q[(ti * 16 + r) * kD + c];
  }
#endif

  // Persistent B fragments for GEMM2 (this wave's 16-channel N-tile).
  Frag b2[2];
#pragma unroll
  for (int ks = 0; ks < 2; ++ks) {
    const unsigned short* p = WT2 + (w * 16 + ln) * kD + ks * 32 + half * 8;
    b2[ks].q4[0] = *(const u32x4*)p;
    b2[ks].q4[1] = *(const u32x4*)(p + 16);
  }

  const int ch = w * 16 + ln;
  float bias2 = bg2[ch];
  float bias1v[4];
#pragma unroll
  for (int nt = 0; nt < 4; ++nt) bias1v[nt] = bg1[nt * 16 + ln];

  // Online softmax state per (query m, channel ch), per lane-half.
  float Mst[16], Lst[16], Ast[16];
#pragma unroll
  for (int m = 0; m < 16; ++m) { Mst[m] = -3.0e30f; Lst[m] = 0.f; Ast[m] = 0.f; }

  for (int tj = 0; tj < 32; ++tj) {
    const int buf = tj & 1;
#if USE_TDM
    if (w == 0) __builtin_amdgcn_s_wait_tensorcnt(0);
#endif
    __syncthreads();                                 // tile tj staged & visible
#if USE_TDM
    if (w == 0 && tj + 1 < 32) {                     // prefetch tile tj+1 via TDM
      tdm_load_tile_f32(lds_off(&sKf[buf ^ 1][0]), k + (tj + 1) * 16 * kD, 16);
      tdm_load_tile_f32(lds_off(&sVf[buf ^ 1][0]), v + (tj + 1) * 16 * kD, 16);
    }
#else
    for (int e = tid; e < 16 * kD; e += 128) {
      int r = e >> 6, c = e & 63;
      sKf[buf][e] = k[(tj * 16 + r) * kD + c];
      sVf[buf][e] = v[(tj * 16 + r) * kD + c];
    }
#endif
    for (int p = tid; p < 256; p += 128) {           // quantized indices
      int ii = p >> 4, jj = p & 15;
      size_t base = ((size_t)(ti * 16 + ii) * kN + (tj * 16 + jj)) * 3;
      float dx = dist[base + 0], dy = dist[base + 1], dz = dist[base + 2];
      int ix = min((int)floorf((dx + 3.0f) * 4.0f), kLXY - 1); ix = max(ix, 0);
      int iy = min((int)floorf((dy + 3.0f) * 4.0f), kLXY - 1); iy = max(iy, 0);
      int iz = min((int)floorf((dz + 2.0f) * 4.0f), kLZ - 1);  iz = max(iz, 0);
      sIDX[p] = (unsigned)ix | ((unsigned)iy << 8) | ((unsigned)iz << 16);
    }
    if (tj + 1 < 32) __builtin_prefetch(&dist[((size_t)ti * 16 * kN + (tj + 1) * 16) * 3], 0, 1);
    __syncthreads();

#pragma unroll
    for (int ph = 0; ph < 2; ++ph) {
      // --- GEMM1: h = relu((q_i - k_j) @ Wg1 + bg1) for 8 m-tiles; 2 per wave.
      Frag b1[4][2];
#pragma unroll
      for (int nt = 0; nt < 4; ++nt)
#pragma unroll
        for (int ks = 0; ks < 2; ++ks) {
          const unsigned short* p = WT1 + (nt * 16 + ln) * kD + ks * 32 + half * 8;
          b1[nt][ks].q4[0] = *(const u32x4*)p;
          b1[nt][ks].q4[1] = *(const u32x4*)(p + 16);
        }
#pragma unroll
      for (int mloc = 0; mloc < 2; ++mloc) {
        const int mt  = ph * 8 + w * 2 + mloc;       // global query index in tile
        const int mtl = w * 2 + mloc;                // local m-tile in sH
        Frag a[2];
#pragma unroll
        for (int ks = 0; ks < 2; ++ks) {
          int kA = ks * 32 + half * 8;
          int kB = kA + 16;
          const float* qr = &sQf[mt * kD];
          const float* kr = &sKf[buf][ln * kD];
          f32x4 qa0 = *(const f32x4*)(qr + kA), qa1 = *(const f32x4*)(qr + kA + 4);
          f32x4 qb0 = *(const f32x4*)(qr + kB), qb1 = *(const f32x4*)(qr + kB + 4);
          f32x4 ka0 = *(const f32x4*)(kr + kA), ka1 = *(const f32x4*)(kr + kA + 4);
          f32x4 kb0 = *(const f32x4*)(kr + kB), kb1 = *(const f32x4*)(kr + kB + 4);
#pragma unroll
          for (int e = 0; e < 4; ++e) {
            a[ks].us[e]      = f2bf(qa0[e] - ka0[e]);
            a[ks].us[4 + e]  = f2bf(qa1[e] - ka1[e]);
            a[ks].us[8 + e]  = f2bf(qb0[e] - kb0[e]);
            a[ks].us[12 + e] = f2bf(qb1[e] - kb1[e]);
          }
        }
#pragma unroll
        for (int nt = 0; nt < 4; ++nt) {
          f32x8 c = {};
          c = __builtin_amdgcn_wmma_f32_16x16x32_bf16(false, a[0].v, false, b1[nt][0].v,
                                                      (short)0, c, false, false);
          c = __builtin_amdgcn_wmma_f32_16x16x32_bf16(false, a[1].v, false, b1[nt][1].v,
                                                      (short)0, c, false, false);
          int n = nt * 16 + ln;
#pragma unroll
          for (int r = 0; r < 8; ++r) {
            int row = mtl * 16 + r + 8 * half;       // C layout: M = vgpr + 8*half
            sH[row * kD + n] = f2bf(fmaxf(c[r] + bias1v[nt], 0.0f));
          }
        }
      }
      __syncthreads();

      // --- GEMM2 + encodings + online per-channel softmax for these 8 m-tiles.
#pragma unroll
      for (int mt2 = 0; mt2 < 8; ++mt2) {
        const int m = ph * 8 + mt2;                  // state index / query in tile
        Frag a[2];
#pragma unroll
        for (int ks = 0; ks < 2; ++ks) {
          const unsigned short* hr = &sH[(mt2 * 16 + ln) * kD + ks * 32 + half * 8];
          a[ks].q4[0] = *(const u32x4*)hr;
          a[ks].q4[1] = *(const u32x4*)(hr + 16);
        }
        f32x8 c = {};
        c = __builtin_amdgcn_wmma_f32_16x16x32_bf16(false, a[0].v, false, b2[0].v,
                                                    (short)0, c, false, false);
        c = __builtin_amdgcn_wmma_f32_16x16x32_bf16(false, a[1].v, false, b2[1].v,
                                                    (short)0, c, false, false);
        float qv = sQf[m * kD + ch];
        float relv[8], vvv[8];
#pragma unroll
        for (int r = 0; r < 8; ++r) {
          int jj = r + 8 * half;
          unsigned pk = sIDX[m * 16 + jj];
          int ix = pk & 255, iy = (pk >> 8) & 255, iz = pk >> 16;
          float eq = bf2f(sTX[(0 * kLXY + ix) * kD + ch]) + bf2f(sTY[(0 * kLXY + iy) * kD + ch]) +
                     bf2f(sTZ[(0 * kLZ + iz) * kD + ch]);
          float ek = bf2f(sTX[(1 * kLXY + ix) * kD + ch]) + bf2f(sTY[(1 * kLXY + iy) * kD + ch]) +
                     bf2f(sTZ[(1 * kLZ + iz) * kD + ch]);
          float ev = bf2f(sTX[(2 * kLXY + ix) * kD + ch]) + bf2f(sTY[(2 * kLXY + iy) * kD + ch]) +
                     bf2f(sTZ[(2 * kLZ + iz) * kD + ch]);
          float kv = sKf[buf][jj * kD + ch];
          relv[r] = c[r] + bias2 + qv * eq + kv * ek;
          vvv[r]  = sVf[buf][jj * kD + ch] + ev;
        }
        float tmax = relv[0];
#pragma unroll
        for (int r = 1; r < 8; ++r) tmax = fmaxf(tmax, relv[r]);
        float nm = fmaxf(Mst[m], tmax);
        float sc = __expf(Mst[m] - nm);
        float ls = 0.f, as = 0.f;
#pragma unroll
        for (int r = 0; r < 8; ++r) {
          float e = __expf(relv[r] - nm);
          ls += e; as += e * vvv[r];
        }
        Lst[m] = Lst[m] * sc + ls;
        Ast[m] = Ast[m] * sc + as;
        Mst[m] = nm;
      }
      __syncthreads();
    }
  }

  // Merge the two lane-half partial softmaxes and write att.
#pragma unroll
  for (int m = 0; m < 16; ++m) {
    float Mo = __shfl_xor(Mst[m], 16, 32);
    float Lo = __shfl_xor(Lst[m], 16, 32);
    float Ao = __shfl_xor(Ast[m], 16, 32);
    float nm = fmaxf(Mst[m], Mo);
    float e0 = __expf(Mst[m] - nm), e1 = __expf(Mo - nm);
    float L = Lst[m] * e0 + Lo * e1;
    float A = Ast[m] * e0 + Ao * e1;
    if (l < 16) att[(ti * 16 + m) * kD + ch] = A / L;
  }
}

// ---------------- K3a: column mean/max + fused constant head term ----------------
__global__ __launch_bounds__(256) void stats_kernel(
    const float* __restrict__ att, const float* __restrict__ Wd,
    const float* __restrict__ bd, float* __restrict__ base) {
  __shared__ float sm[256], sx[256];
  int t = threadIdx.x;
  int c = t & 63, s = t >> 6;
  float sum = 0.f, mx = -3.0e30f;
  for (int i = s; i < kN; i += 4) {
    float vv = att[i * kD + c];
    sum += vv; mx = fmaxf(mx, vv);
  }
  sm[t] = sum; sx[t] = mx;
  __syncthreads();
  if (t < 64) {
    sm[t] = (sm[t] + sm[t + 64] + sm[t + 128] + sm[t + 192]) * (1.0f / kN);
    sx[t] = fmaxf(fmaxf(sx[t], sx[t + 64]), fmaxf(sx[t + 128], sx[t + 192]));
  }
  __syncthreads();
  if (t < 64) {
    float acc = bd[t];
    for (int c2 = 0; c2 < 64; ++c2) {
      acc = fmaf(sm[c2], Wd[(64 + c2) * kD + t], acc);
      acc = fmaf(sx[c2], Wd[(128 + c2) * kD + t], acc);
    }
    base[t] = acc;
  }
}

// ---------------- K3b: per-row GEMV + LayerNorm + relu ----------------
__global__ __launch_bounds__(64) void head_kernel(
    const float* __restrict__ att, const float* __restrict__ Wd,
    const float* __restrict__ base, const float* __restrict__ lng,
    const float* __restrict__ lnb, float* __restrict__ out) {
  __shared__ float row[64], red[64];
  int i = blockIdx.x, t = threadIdx.x;
  row[t] = att[i * kD + t];
  __syncthreads();
  float h = base[t];
#pragma unroll 8
  for (int kk = 0; kk < 64; ++kk) h = fmaf(row[kk], Wd[kk * kD + t], h);
  red[t] = h; __syncthreads();
  for (int off = 32; off > 0; off >>= 1) { if (t < off) red[t] += red[t + off]; __syncthreads(); }
  float mu = red[0] * (1.0f / 64); __syncthreads();
  float d = h - mu;
  red[t] = d * d; __syncthreads();
  for (int off = 32; off > 0; off >>= 1) { if (t < off) red[t] += red[t + off]; __syncthreads(); }
  float var = red[0] * (1.0f / 64);
  out[i * kD + t] = fmaxf(d * rsqrtf(var + 1e-5f) * lng[t] + lnb[t], 0.0f);
}

extern "C" void kernel_launch(void* const* d_in, const int* in_sizes, int n_in,
                              void* d_out, int out_size, void* d_ws, size_t ws_size,
                              hipStream_t stream) {
  const float* feat = (const float*)d_in[0];
  const float* dist = (const float*)d_in[1];
  const float* Wqkv = (const float*)d_in[2];
  const float* bqkv = (const float*)d_in[3];
  const float* Wg1  = (const float*)d_in[4];
  const float* bg1  = (const float*)d_in[5];
  const float* Wg2  = (const float*)d_in[6];
  const float* bg2  = (const float*)d_in[7];
  const float* tx   = (const float*)d_in[8];
  const float* ty   = (const float*)d_in[9];
  const float* tz   = (const float*)d_in[10];
  const float* Wd   = (const float*)d_in[11];
  const float* bd   = (const float*)d_in[12];
  const float* lng  = (const float*)d_in[13];
  const float* lnb  = (const float*)d_in[14];

  // Workspace layout (~542 KB): q,k,v,att (f32) | base (f32) | WT1,WT2 (bf16)
  float* q    = (float*)d_ws;
  float* k    = q + kN * kD;
  float* v    = k + kN * kD;
  float* att  = v + kN * kD;
  float* base = att + kN * kD;
  unsigned short* WT1 = (unsigned short*)(base + 64);   // 16B-aligned offset
  unsigned short* WT2 = WT1 + kD * kD;

  qkv_kernel  <<<(kN * 192) / 256, 256, 0, stream>>>(feat, Wqkv, bqkv, q, k, v);
  prep_weights<<<1, 256, 0, stream>>>(Wg1, Wg2, WT1, WT2);
  vec_att_kernel<<<kN / 16, 128, 0, stream>>>(q, k, v, dist, bg1, bg2,
                                              WT1, WT2, tx, ty, tz, att);
  stats_kernel<<<1, 256, 0, stream>>>(att, Wd, bd, base);
  head_kernel <<<kN, 64, 0, stream>>>(att, Wd, base, lng, lnb, (float*)d_out);

  (void)in_sizes; (void)n_in; (void)out_size; (void)ws_size;
}
